// MDTA_58789512348003
// MI455X (gfx1250) — matrix operand order repaired and checked by reading, not compiled, not run on hardware
//
#include <hip/hip_runtime.h>
#include <hip/hip_bf16.h>
#include <math.h>

// ---------------------------------------------------------------------------
// MDTA (Restormer multi-dconv-head transposed attention) for MI455X (gfx1250)
// b=8, c=192, h=w=128 (hw=16384), heads=4, dh=48
// Matrix stages: v_wmma_f32_16x16x32_bf16 (wave32 WMMA).
// LDS staging:   global_load_async_to_lds_b128, double-buffered with partial
//                s_wait_asynccnt waits (ASYNCcnt) to overlap DMA with WMMA.
// ---------------------------------------------------------------------------

typedef __bf16 bf16_t;
typedef __bf16 v16bf __attribute__((ext_vector_type(16)));
typedef float  v8f   __attribute__((ext_vector_type(8)));

#define HW   16384
#define HDIM 128
#define C    192
#define C3   576
#define B    8
#define HEADS 4
#define DH   48

static __device__ __forceinline__ bf16_t f2bf(float f) { return (bf16_t)f; }
static __device__ __forceinline__ float  bf2f(bf16_t b) { return (float)b; }

#define LDS_AS __attribute__((address_space(3)))

static __device__ __forceinline__ unsigned ldsoff(const void* p) {
    return (unsigned)(unsigned long long)(const LDS_AS char*)p;
}

// Async DMA: 16 bytes global -> LDS, tracked by ASYNCcnt.
static __device__ __forceinline__ void async_b128(void* ldst, const void* g) {
#if defined(__gfx1250__)
    asm volatile("global_load_async_to_lds_b128 %0, %1, off"
                 :: "v"(ldsoff(ldst)), "v"((unsigned long long)g)
                 : "memory");
#else
    ((unsigned long long*)ldst)[0] = ((const unsigned long long*)g)[0];
    ((unsigned long long*)ldst)[1] = ((const unsigned long long*)g)[1];
#endif
}

static __device__ __forceinline__ void wait_async0() {  // all complete
#if defined(__gfx1250__)
    asm volatile("s_wait_asynccnt 0" ::: "memory");
#endif
}
static __device__ __forceinline__ void wait_async2() {  // allow 2 in flight
#if defined(__gfx1250__)
    asm volatile("s_wait_asynccnt 2" ::: "memory");
#endif
}

// 16-bit A-matrix (16x32) K index for fragment element e, given lane-half.
static __device__ __forceinline__ int kmapA(int e, int half) {
    int j = e >> 1, pos = e & 1;
    return ((j < 4) ? 0 : 16) + half * 8 + ((j & 3) << 1) + pos;
}
// 16-bit B-matrix (32x16): same K order as A
static __device__ __forceinline__ int kmapB(int e, int half) {
    return ((e < 8) ? 0 : 8) + half * 8 + e;
}

static __device__ __forceinline__ v8f wmma_bf16(v16bf a, v16bf b, v8f c) {
    return __builtin_amdgcn_wmma_f32_16x16x32_bf16(false, a, false, b,
                                                   (short)0, c, false, false);
}

// ---------------------------------------------------------------------------
// Kernel 1/5: channel GEMM (1x1 conv).  Out[b][o][n] = sum_k W[o][k]*X[b][k][n]
// grid = (HW/128, M/64, B), block = 256 (8 waves). Tile 64(M) x 128(N).
// ---------------------------------------------------------------------------
template <bool IN_BF16, bool OUT_F32>
__global__ __launch_bounds__(256)
void gemm1x1_kernel(const void* __restrict__ Xv, const float* __restrict__ W,
                    void* __restrict__ Outv, int M, int K) {
    __shared__ __align__(16) bf16_t lds_w[64][32];
    __shared__ __align__(16) bf16_t lds_x[32][128];

    const int tid  = threadIdx.x;
    const int lane = tid & 31;
    const int wave = tid >> 5;
    const int col  = lane & 15;
    const int half = lane >> 4;
    const int mi   = wave >> 1;
    const int nh   = wave & 1;

    const int n0    = blockIdx.x * 128;
    const int m0    = blockIdx.y * 64;
    const int batch = blockIdx.z;

    const float*  Xf = (const float*)Xv;
    const bf16_t* Xb = (const bf16_t*)Xv;
    const size_t  xbase = (size_t)batch * K * HW;

    v8f acc[4];
#pragma unroll
    for (int s = 0; s < 4; ++s)
#pragma unroll
        for (int r = 0; r < 8; ++r) acc[s][r] = 0.0f;

    for (int k0 = 0; k0 < K; k0 += 32) {
        // stage W tile [64][32]: fp32 -> packed bf16x4
#pragma unroll
        for (int i = 0; i < 2; ++i) {
            int e4 = tid + i * 256;
            int r = e4 >> 3, c4 = e4 & 7;
            float4 wv = *(const float4*)&W[(size_t)(m0 + r) * K + k0 + c4 * 4];
            union { bf16_t h[4]; unsigned long long u; } pk;
            pk.h[0] = f2bf(wv.x); pk.h[1] = f2bf(wv.y);
            pk.h[2] = f2bf(wv.z); pk.h[3] = f2bf(wv.w);
            *(unsigned long long*)&lds_w[r][c4 * 4] = pk.u;
        }
        // stage X tile [32][128]
        if (IN_BF16) {
#pragma unroll
            for (int i = 0; i < 2; ++i) {
                int e = tid + i * 256;
                int r = e >> 4, ck = e & 15;
                async_b128(&lds_x[r][ck * 8],
                           Xb + xbase + (size_t)(k0 + r) * HW + n0 + ck * 8);
            }
            wait_async0();
        } else {
#pragma unroll
            for (int i = 0; i < 4; ++i) {
                int e4 = tid + i * 256;
                int r = e4 >> 5, c4 = e4 & 31;
                float4 xv = *(const float4*)&Xf[xbase + (size_t)(k0 + r) * HW + n0 + c4 * 4];
                union { bf16_t h[4]; unsigned long long u; } pk;
                pk.h[0] = f2bf(xv.x); pk.h[1] = f2bf(xv.y);
                pk.h[2] = f2bf(xv.z); pk.h[3] = f2bf(xv.w);
                *(unsigned long long*)&lds_x[r][c4 * 4] = pk.u;
            }
            if (k0 + 32 < K)
                __builtin_prefetch((const char*)&Xf[xbase + (size_t)(k0 + 32) * HW + n0]
                                   + (tid & 15) * 64, 0, 0);
        }
        __syncthreads();

        v16bf a;
#pragma unroll
        for (int e = 0; e < 16; ++e)
            a[e] = lds_w[mi * 16 + col][kmapA(e, half)];
#pragma unroll
        for (int s = 0; s < 4; ++s) {
            v16bf b;
#pragma unroll
            for (int e = 0; e < 16; ++e)
                b[e] = lds_x[kmapB(e, half)][nh * 64 + s * 16 + col];
            acc[s] = wmma_bf16(a, b, acc[s]);
        }
        __syncthreads();
    }

    float*  Of = (float*)Outv;
    bf16_t* Ob = (bf16_t*)Outv;
#pragma unroll
    for (int s = 0; s < 4; ++s) {
#pragma unroll
        for (int r = 0; r < 8; ++r) {
            int row = m0 + mi * 16 + r + half * 8;
            int nc  = n0 + nh * 64 + s * 16 + col;
            size_t off = (size_t)batch * M * HW + (size_t)row * HW + nc;
            if (OUT_F32) Of[off] = acc[s][r];
            else         Ob[off] = f2bf(acc[s][r]);
        }
    }
}

// ---------------------------------------------------------------------------
// Kernel 2/5: 3x3 depthwise conv (padding=1) on bf16 qkv.
// ---------------------------------------------------------------------------
__global__ __launch_bounds__(256)
void dwconv_kernel(const bf16_t* __restrict__ in, const float* __restrict__ wdw,
                   bf16_t* __restrict__ out) {
    const int ch  = blockIdx.y;
    const int bi  = blockIdx.z;
    const int pix = blockIdx.x * 256 + threadIdx.x;
    const int y   = pix >> 7;
    const int x   = pix & 127;
    const size_t base = ((size_t)bi * C3 + ch) * HW;

    float accv = 0.0f;
#pragma unroll
    for (int dy = -1; dy <= 1; ++dy) {
#pragma unroll
        for (int dx = -1; dx <= 1; ++dx) {
            int yy = y + dy, xx = x + dx;
            if (yy >= 0 && yy < HDIM && xx >= 0 && xx < HDIM) {
                accv += bf2f(in[base + yy * HDIM + xx]) *
                        wdw[ch * 9 + (dy + 1) * 3 + (dx + 1)];
            }
        }
    }
    out[base + pix] = f2bf(accv);
}

// ---------------------------------------------------------------------------
// Kernel 3/5: inverse L2 norm along hw for q and k channels.
// ---------------------------------------------------------------------------
__global__ __launch_bounds__(256)
void invnorm_kernel(const bf16_t* __restrict__ qk, float* __restrict__ inv) {
    __shared__ float red[256];
    const int ch = blockIdx.x;
    const int bi = blockIdx.y;
    const size_t base = ((size_t)bi * C3 + ch) * HW;

    float s = 0.0f;
    for (int i = threadIdx.x; i < HW; i += 256) {
        float v = bf2f(qk[base + i]);
        s += v * v;
    }
    red[threadIdx.x] = s;
    __syncthreads();
#pragma unroll
    for (int d = 128; d > 0; d >>= 1) {
        if (threadIdx.x < d) red[threadIdx.x] += red[threadIdx.x + d];
        __syncthreads();
    }
    if (threadIdx.x == 0)
        inv[(size_t)bi * (2 * C) + ch] = 1.0f / fmaxf(sqrtf(red[0]), 1e-12f);
}

// ---------------------------------------------------------------------------
// Kernel 4/5: fused transposed attention per (b, head). block = 384 (12 waves)
// Phase 1: S = q.k^T over K=16384, double-buffered 48x64 q/k tiles (async DMA
//          for tile t+1 in flight while tile t is in WMMA; s_wait_asynccnt 2).
// Phase 2: out = attn.v, double-buffered 64x128 v tiles, same pipeline.
// LDS overlay (32KB region): q/k bufs -> sattn -> v bufs. Total 39.3KB.
// ---------------------------------------------------------------------------
__global__ __launch_bounds__(384)
void attention_kernel(const bf16_t* __restrict__ qkv,
                      const float* __restrict__ inv,
                      const float* __restrict__ temperature,
                      bf16_t* __restrict__ ao) {
    __shared__ __align__(16) char smem[32768 + 6144 + 384];
    // dynamic LDS buffer addressing (no static-initialized pointer tables)
#define LQ(pp) ((bf16_t(*)[64])(smem + (pp) * 6144))
#define LK(pp) ((bf16_t(*)[64])(smem + 12288 + (pp) * 6144))
#define LV(pp) ((bf16_t(*)[128])(smem + (pp) * 16384))
    float  (*sattn)[DH + 1] = (float(*)[DH + 1])(smem);     // lives after phase 1
    bf16_t (*attn_bf)[64]   = (bf16_t(*)[64])(smem + 32768);
    float*  s_invq          = (float*)(smem + 32768 + 6144);
    float*  s_invk          = s_invq + DH;

    const int head = blockIdx.x;
    const int bi   = blockIdx.y;
    const int tid  = threadIdx.x;
    const int lane = tid & 31;
    const int wave = tid >> 5;
    const int col  = lane & 15;
    const int half = lane >> 4;

    const size_t qbase = ((size_t)bi * C3 + head * DH) * HW;
    const size_t kbase = ((size_t)bi * C3 + C + head * DH) * HW;
    const size_t vbase = ((size_t)bi * C3 + 2 * C + head * DH) * HW;

    if (tid < DH)
        s_invq[tid] = inv[(size_t)bi * (2 * C) + head * DH + tid];
    else if (tid < 2 * DH)
        s_invk[tid - DH] = inv[(size_t)bi * (2 * C) + C + head * DH + (tid - DH)];
    const float temp = temperature[head];

    // issue one 48x64 q tile + k tile (2 async instr/thread: 384 chunks each)
    auto issue_qk = [&](int k0, int pp) {
        int row = tid >> 3, ck = tid & 7;
        async_b128(&LQ(pp)[row][ck * 8],
                   qkv + qbase + (size_t)row * HW + k0 + ck * 8);
        async_b128(&LK(pp)[row][ck * 8],
                   qkv + kbase + (size_t)row * HW + k0 + ck * 8);
    };
    // issue one 48x128 v tile (2 async instr/thread: 768 chunks)
    auto issue_v = [&](int n0, int pp) {
#pragma unroll
        for (int i = 0; i < 2; ++i) {
            int c = tid + i * 384;
            int row = c >> 4, ck = c & 15;
            async_b128(&LV(pp)[row][ck * 8],
                       qkv + vbase + (size_t)row * HW + n0 + ck * 8);
        }
    };

    // ---- Phase 1: Gram matrix over n, double-buffered K-tile = 64 ----
    const int mi = wave / 3, ni = wave % 3;   // waves 0..8 own 3x3 C tiles
    v8f acc;
#pragma unroll
    for (int r = 0; r < 8; ++r) acc[r] = 0.0f;

    int p = 0;
    issue_qk(0, 0);
    for (int k0 = 0; k0 < HW; k0 += 64) {
        if (k0 + 64 < HW) { issue_qk(k0 + 64, p ^ 1); wait_async2(); }
        else              { wait_async0(); }
        __syncthreads();
        if (wave < 9) {
#pragma unroll
            for (int kk = 0; kk < 2; ++kk) {
                v16bf a, b;
#pragma unroll
                for (int e = 0; e < 16; ++e) {
                    a[e] = LQ(p)[mi * 16 + col][kk * 32 + kmapA(e, half)];
                    b[e] = LK(p)[ni * 16 + col][kk * 32 + kmapB(e, half)];
                }
                acc = wmma_bf16(a, b, acc);
            }
        }
        __syncthreads();
        p ^= 1;
    }
    if (wave < 9) {
#pragma unroll
        for (int r = 0; r < 8; ++r) {
            int m = mi * 16 + r + half * 8;
            int d = ni * 16 + col;
            sattn[m][d] = acc[r] * s_invq[m] * s_invk[d] * temp;
        }
    }
    __syncthreads();

    // ---- softmax over d ----
    if (tid < DH) {
        int row = tid;
        float mx = -1e30f;
        for (int d = 0; d < DH; ++d) mx = fmaxf(mx, sattn[row][d]);
        float sum = 0.0f;
        for (int d = 0; d < DH; ++d) {
            float e = __expf(sattn[row][d] - mx);
            sattn[row][d] = e;
            sum += e;
        }
        float rs = 1.0f / sum;
        for (int d = 0; d < 64; ++d)
            attn_bf[row][d] = (d < DH) ? f2bf(sattn[row][d] * rs) : f2bf(0.0f);
    }
    __syncthreads();   // sattn dead beyond this point; v buffers may overwrite

    // zero pad rows 48..63 of both v buffers once (never overwritten below)
    for (int c = tid; c < 512; c += 384) {  // 2 bufs x 16 rows x 16 chunks
        int buf = c >> 8, cc = c & 255;
        int row = 48 + (cc >> 4), ck = cc & 15;
        ((unsigned long long*)&LV(buf)[row][ck * 8])[0] = 0ULL;
        ((unsigned long long*)&LV(buf)[row][ck * 8])[1] = 0ULL;
    }

    // ---- Phase 2: out = attn(48x48, K padded to 64) . v, double-buffered ----
    const int mi2 = wave >> 2;   // 0..2
    const int nt  = wave & 3;    // 0..3 -> 32 columns each
    p = 0;
    issue_v(0, 0);
    for (int n0 = 0; n0 < HW; n0 += 128) {
        if (n0 + 128 < HW) { issue_v(n0 + 128, p ^ 1); wait_async2(); }
        else               { wait_async0(); }
        __syncthreads();

        v8f oacc[2];
#pragma unroll
        for (int s = 0; s < 2; ++s)
#pragma unroll
            for (int r = 0; r < 8; ++r) oacc[s][r] = 0.0f;

#pragma unroll
        for (int dd0 = 0; dd0 < 64; dd0 += 32) {
            v16bf a;
#pragma unroll
            for (int e = 0; e < 16; ++e)
                a[e] = attn_bf[mi2 * 16 + col][dd0 + kmapA(e, half)];
#pragma unroll
            for (int s = 0; s < 2; ++s) {
                v16bf b;
#pragma unroll
                for (int e = 0; e < 16; ++e)
                    b[e] = LV(p)[dd0 + kmapB(e, half)][nt * 32 + s * 16 + col];
                oacc[s] = wmma_bf16(a, b, oacc[s]);
            }
        }
#pragma unroll
        for (int s = 0; s < 2; ++s) {
#pragma unroll
            for (int r = 0; r < 8; ++r) {
                int cout = head * DH + mi2 * 16 + r + half * 8;
                int nc   = n0 + nt * 32 + s * 16 + col;
                ao[((size_t)bi * C + cout) * HW + nc] = f2bf(oacc[s][r]);
            }
        }
        __syncthreads();
        p ^= 1;
    }
#undef LQ
#undef LK
#undef LV
}

// ---------------------------------------------------------------------------
extern "C" void kernel_launch(void* const* d_in, const int* in_sizes, int n_in,
                              void* d_out, int out_size, void* d_ws, size_t ws_size,
                              hipStream_t stream) {
    const float* x      = (const float*)d_in[0];  // [8,192,128,128]
    const float* w_qkv  = (const float*)d_in[1];  // [576,192]
    const float* w_dw   = (const float*)d_in[2];  // [576,1,3,3]
    const float* w_proj = (const float*)d_in[3];  // [192,192]
    const float* temp   = (const float*)d_in[4];  // [4]
    (void)in_sizes; (void)n_in; (void)out_size; (void)ws_size;

    const size_t S_QKV = (size_t)B * C3 * HW * sizeof(bf16_t);  // 150,994,944
    char* ws = (char*)d_ws;
    bf16_t* qkv_a  = (bf16_t*)(ws);                         // post 1x1 conv
    bf16_t* qkv_dw = (bf16_t*)(ws + S_QKV);                 // post depthwise
    float*  invn   = (float*)(ws + 2 * S_QKV);              // [B][2C]
    bf16_t* ao     = (bf16_t*)(ws + 2 * S_QKV + 65536);     // attn out [B][C][HW]

    gemm1x1_kernel<false, false><<<dim3(HW / 128, C3 / 64, B), 256, 0, stream>>>(
        (const void*)x, w_qkv, (void*)qkv_a, C3, C);

    dwconv_kernel<<<dim3(HW / 256, C3, B), 256, 0, stream>>>(qkv_a, w_dw, qkv_dw);

    invnorm_kernel<<<dim3(2 * C, B), 256, 0, stream>>>(qkv_dw, invn);

    attention_kernel<<<dim3(HEADS, B), 384, 0, stream>>>(qkv_dw, invn, temp, ao);

    gemm1x1_kernel<true, true><<<dim3(HW / 128, C / 64, B), 256, 0, stream>>>(
        (const void*)ao, w_proj, d_out, C, C);
}
